// IxformerQuantMoe_80908593922370
// MI455X (gfx1250) — 8-bit matrix operand order repaired and checked by reading, compiled, not run on hardware
//
#include <hip/hip_runtime.h>
#include <hip/hip_bf16.h>

#define T_TOK 8192
#define H_DIM 2048
#define I_DIM 1408
#define N1_DIM 2816
#define E_EXP 8

typedef __attribute__((ext_vector_type(8))) int v8i;

__device__ __forceinline__ int qpack4(float a, float b, float c, float d, float inv) {
  int ia = (int)fminf(fmaxf(rintf(a * inv), -127.f), 127.f);
  int ib = (int)fminf(fmaxf(rintf(b * inv), -127.f), 127.f);
  int ic = (int)fminf(fmaxf(rintf(c * inv), -127.f), 127.f);
  int id = (int)fminf(fmaxf(rintf(d * inv), -127.f), 127.f);
  return (ia & 0xff) | ((ib & 0xff) << 8) | ((ic & 0xff) << 16) | ((id & 0xff) << 24);
}

__global__ void pack_i32_i8_kernel(const int* __restrict__ src, int* __restrict__ dst, int n4) {
  int i = blockIdx.x * blockDim.x + threadIdx.x;
  if (i >= n4) return;
  int4 v = ((const int4*)src)[i];
  dst[i] = (v.x & 0xff) | ((v.y & 0xff) << 8) | ((v.z & 0xff) << 16) | ((v.w & 0xff) << 24);
}

__global__ void zero_cnt_kernel(int* cnt) {
  if (threadIdx.x < E_EXP) cnt[threadIdx.x] = 0;
}

__global__ void gate_topk_kernel(const float* __restrict__ x, const float* __restrict__ gw,
                                 int* __restrict__ cnt, int* __restrict__ tok_list,
                                 float* __restrict__ gate_list) {
  int wave = (blockIdx.x * blockDim.x + threadIdx.x) >> 5;
  int lane = threadIdx.x & 31;
  if (wave >= T_TOK) return;
  const float* xr = x + (size_t)wave * H_DIM;
  float acc[E_EXP];
#pragma unroll
  for (int e = 0; e < E_EXP; e++) acc[e] = 0.f;
  for (int h = lane; h < H_DIM; h += 32) {
    float xv = xr[h];
#pragma unroll
    for (int e = 0; e < E_EXP; e++) acc[e] += xv * gw[e * H_DIM + h];
  }
#pragma unroll
  for (int e = 0; e < E_EXP; e++)
#pragma unroll
    for (int off = 16; off >= 1; off >>= 1) acc[e] += __shfl_xor(acc[e], off, 32);
  if (lane == 0) {
    float m = acc[0];
#pragma unroll
    for (int e = 1; e < E_EXP; e++) m = fmaxf(m, acc[e]);
    float p[E_EXP];
#pragma unroll
    for (int e = 0; e < E_EXP; e++) p[e] = expf(acc[e] - m);
    int i0 = 0;
#pragma unroll
    for (int e = 1; e < E_EXP; e++) if (p[e] > p[i0]) i0 = e;
    int i1 = (i0 == 0) ? 1 : 0;
#pragma unroll
    for (int e = 0; e < E_EXP; e++) if (e != i0 && p[e] > p[i1]) i1 = e;
    float g0 = p[i0], g1 = p[i1];
    float gs = g0 + g1;
    g0 /= gs; g1 /= gs;
    int s0 = atomicAdd(&cnt[i0], 1);
    tok_list[i0 * T_TOK + s0] = wave;
    gate_list[i0 * T_TOK + s0] = g0;
    int s1 = atomicAdd(&cnt[i1], 1);
    tok_list[i1 * T_TOK + s1] = wave;
    gate_list[i1 * T_TOK + s1] = g1;
  }
}

__global__ void quant_x_kernel(const float* __restrict__ x, char* __restrict__ xq,
                               float* __restrict__ xs) {
  int wave = (blockIdx.x * blockDim.x + threadIdx.x) >> 5;
  int lane = threadIdx.x & 31;
  if (wave >= T_TOK) return;
  const float4* xr = (const float4*)(x + (size_t)wave * H_DIM);
  float amax = 0.f;
  for (int i = lane; i < H_DIM / 4; i += 32) {
    float4 v = xr[i];
    amax = fmaxf(amax, fmaxf(fmaxf(fabsf(v.x), fabsf(v.y)), fmaxf(fabsf(v.z), fabsf(v.w))));
  }
#pragma unroll
  for (int off = 16; off >= 1; off >>= 1) amax = fmaxf(amax, __shfl_xor(amax, off, 32));
  float s = fmaxf(amax / 127.0f, 1e-8f);
  if (lane == 0) xs[wave] = s;
  float inv = 1.0f / s;
  int* q = (int*)(xq + (size_t)wave * H_DIM);
  for (int i = lane; i < H_DIM / 4; i += 32) {
    float4 v = xr[i];
    q[i] = qpack4(v.x, v.y, v.z, v.w, inv);
  }
}


__global__ __launch_bounds__(256)
void fc1_kernel(const char* __restrict__ xq, const float* __restrict__ xs,
                const char* __restrict__ w13p, const float* __restrict__ w13_scale,
                const int* __restrict__ cnt, const int* __restrict__ tok_list,
                float* __restrict__ h_buf, int e) {
  __shared__ __align__(16) char As[128 * 128];
  __shared__ __align__(16) char Bs[64 * 128];
  __shared__ int   toks[128];
  __shared__ float xss[128];
  int count = cnt[e];
  int m0 = blockIdx.y * 128;
  if (m0 >= count) return;
  int n0 = blockIdx.x * 64;
  int tid = threadIdx.x;
  if (tid < 128) {
    int slot = m0 + tid;
    int tk = (slot < count) ? tok_list[e * T_TOK + slot] : -1;
    toks[tid] = tk;
    xss[tid] = (tk >= 0) ? xs[tk] : 0.f;
  }
  __syncthreads();
  int wv = tid >> 5, lane = tid & 31;
  int mwp = wv & 3;
  int nwp = wv >> 2;
  v8i acc[2][2];
#pragma unroll
  for (int im = 0; im < 2; im++)
#pragma unroll
    for (int in = 0; in < 2; in++) acc[im][in] = (v8i){0, 0, 0, 0, 0, 0, 0, 0};

  const size_t wbase = (size_t)e * N1_DIM * H_DIM + (size_t)n0 * H_DIM;
  int hi = (lane >> 4) & 1;
  int lane15 = lane & 15;
  int hi8 = hi * 8;

  for (int kk = 0; kk < H_DIM; kk += 128) {
#pragma unroll
    for (int t = 0; t < 4; t++) {
      int c = tid + t * 256;
      int row = c >> 3;
      int col = (c & 7) * 16;
      int4 av = {0, 0, 0, 0};
      int tk = toks[row];
      if (tk >= 0) {
        const char* gp = xq + (size_t)tk * H_DIM + kk + col;
        av = *(const int4*)gp;
        __builtin_prefetch(gp + 128, 0, 1);
      }
      *(int4*)(As + row * 128 + col) = av;
    }
#pragma unroll
    for (int t = 0; t < 2; t++) {
      int c = tid + t * 256;
      int row = c >> 3;
      int col = (c & 7) * 16;
      const char* gp = w13p + wbase + (size_t)row * H_DIM + kk + col;
      *(int4*)(Bs + row * 128 + col) = *(const int4*)gp;
      __builtin_prefetch(gp + 128, 0, 1);
    }
    __syncthreads();
#pragma unroll
    for (int s = 0; s < 2; s++) {
      v8i a[2], b[2];
#pragma unroll
      for (int im = 0; im < 2; im++) {
        int arow = (mwp * 2 + im) * 16 + lane15;
#pragma unroll
        for (int v = 0; v < 8; v++) {
          int aoff = ((v >> 1) & 1) * 16 + (v & 1) * 4 + (v >> 2) * 32 + hi * 8;
          a[im][v] = *(const int*)(As + arow * 128 + s * 64 + aoff);
        }
      }
#pragma unroll
      for (int in = 0; in < 2; in++) {
        int brow = (nwp * 2 + in) * 16 + lane15;
#pragma unroll
        for (int v = 0; v < 8; v++) {
          int boff = ((v >> 1) & 1) * 16 + (v & 1) * 4 + (v >> 2) * 32 + hi * 8;
          b[in][v] = *(const int*)(Bs + brow * 128 + s * 64 + boff);
        }
      }
#pragma unroll
      for (int im = 0; im < 2; im++)
#pragma unroll
        for (int in = 0; in < 2; in++)
          acc[im][in] = __builtin_amdgcn_wmma_i32_16x16x64_iu8(
              true, a[im], true, b[in], acc[im][in], false, false);
    }
    __syncthreads();
  }
#pragma unroll
  for (int im = 0; im < 2; im++) {
#pragma unroll
    for (int in = 0; in < 2; in++) {
      int ng = n0 + (nwp * 2 + in) * 16 + lane15;
      float wsc = w13_scale[e * N1_DIM + ng];
#pragma unroll
      for (int r = 0; r < 8; r++) {
        int mloc = (mwp * 2 + im) * 16 + r + hi8;
        int slot = m0 + mloc;
        if (slot < count) {
          float val = (float)acc[im][in][r] * xss[mloc] * wsc;
          h_buf[(size_t)slot * N1_DIM + ng] = val;
        }
      }
    }
  }
}

__global__ void swiglu_quant_kernel(const float* __restrict__ h_buf, char* __restrict__ aq,
                                    float* __restrict__ as_, const int* __restrict__ cnt, int e) {
  int wave = (blockIdx.x * blockDim.x + threadIdx.x) >> 5;
  int lane = threadIdx.x & 31;
  int count = cnt[e];
  if (wave >= count) return;
  const float4* g4 = (const float4*)(h_buf + (size_t)wave * N1_DIM);
  const float4* u4 = (const float4*)(h_buf + (size_t)wave * N1_DIM + I_DIM);
  float amax = 0.f;
  for (int i = lane; i < I_DIM / 4; i += 32) {
    float4 g = g4[i], u = u4[i];
    float ax = g.x / (1.f + expf(-g.x)) * u.x;
    float ay = g.y / (1.f + expf(-g.y)) * u.y;
    float az = g.z / (1.f + expf(-g.z)) * u.z;
    float aw = g.w / (1.f + expf(-g.w)) * u.w;
    amax = fmaxf(amax, fmaxf(fmaxf(fabsf(ax), fabsf(ay)), fmaxf(fabsf(az), fabsf(aw))));
  }
#pragma unroll
  for (int off = 16; off >= 1; off >>= 1) amax = fmaxf(amax, __shfl_xor(amax, off, 32));
  float s = fmaxf(amax / 127.0f, 1e-8f);
  if (lane == 0) as_[wave] = s;
  float inv = 1.0f / s;
  int* q = (int*)(aq + (size_t)wave * I_DIM);
  for (int i = lane; i < I_DIM / 4; i += 32) {
    float4 g = g4[i], u = u4[i];
    float ax = g.x / (1.f + expf(-g.x)) * u.x;
    float ay = g.y / (1.f + expf(-g.y)) * u.y;
    float az = g.z / (1.f + expf(-g.z)) * u.z;
    float aw = g.w / (1.f + expf(-g.w)) * u.w;
    q[i] = qpack4(ax, ay, az, aw, inv);
  }
}

__global__ __launch_bounds__(256)
void fc2_kernel(const char* __restrict__ aq, const float* __restrict__ as_,
                const char* __restrict__ w2p, const float* __restrict__ w2_scale,
                const int* __restrict__ cnt, const int* __restrict__ tok_list,
                const float* __restrict__ gate_list, float* __restrict__ out, int e) {
  __shared__ __align__(16) char As[128 * 128];
  __shared__ __align__(16) char Bs[64 * 128];
  __shared__ int   toks[128];
  __shared__ float gl[128];
  __shared__ float ass[128];
  int count = cnt[e];
  int m0 = blockIdx.y * 128;
  if (m0 >= count) return;
  int n0 = blockIdx.x * 64;
  int tid = threadIdx.x;
  if (tid < 128) {
    int slot = m0 + tid;
    bool ok = (slot < count);
    toks[tid] = ok ? tok_list[e * T_TOK + slot] : -1;
    gl[tid]   = ok ? gate_list[e * T_TOK + slot] : 0.f;
    ass[tid]  = ok ? as_[slot] : 0.f;
  }
  __syncthreads();
  int wv = tid >> 5, lane = tid & 31;
  int mwp = wv & 3;
  int nwp = wv >> 2;
  v8i acc[2][2];
#pragma unroll
  for (int im = 0; im < 2; im++)
#pragma unroll
    for (int in = 0; in < 2; in++) acc[im][in] = (v8i){0, 0, 0, 0, 0, 0, 0, 0};

  const size_t wbase = (size_t)e * H_DIM * I_DIM + (size_t)n0 * I_DIM;
  int hi = (lane >> 4) & 1;
  int lane15 = lane & 15;
  int hi8 = hi * 8;

  for (int kk = 0; kk < I_DIM; kk += 128) {
#pragma unroll
    for (int t = 0; t < 4; t++) {
      int c = tid + t * 256;
      int row = c >> 3;
      int col = (c & 7) * 16;
      int slot = m0 + row;
      int4 av = {0, 0, 0, 0};
      if (slot < count) {
        const char* gp = aq + (size_t)slot * I_DIM + kk + col;
        av = *(const int4*)gp;
        __builtin_prefetch(gp + 128, 0, 1);
      }
      *(int4*)(As + row * 128 + col) = av;
    }
#pragma unroll
    for (int t = 0; t < 2; t++) {
      int c = tid + t * 256;
      int row = c >> 3;
      int col = (c & 7) * 16;
      const char* gp = w2p + wbase + (size_t)row * I_DIM + kk + col;
      *(int4*)(Bs + row * 128 + col) = *(const int4*)gp;
      __builtin_prefetch(gp + 128, 0, 1);
    }
    __syncthreads();
#pragma unroll
    for (int s = 0; s < 2; s++) {
      v8i a[2], b[2];
#pragma unroll
      for (int im = 0; im < 2; im++) {
        int arow = (mwp * 2 + im) * 16 + lane15;
#pragma unroll
        for (int v = 0; v < 8; v++) {
          int aoff = ((v >> 1) & 1) * 16 + (v & 1) * 4 + (v >> 2) * 32 + hi * 8;
          a[im][v] = *(const int*)(As + arow * 128 + s * 64 + aoff);
        }
      }
#pragma unroll
      for (int in = 0; in < 2; in++) {
        int brow = (nwp * 2 + in) * 16 + lane15;
#pragma unroll
        for (int v = 0; v < 8; v++) {
          int boff = ((v >> 1) & 1) * 16 + (v & 1) * 4 + (v >> 2) * 32 + hi * 8;
          b[in][v] = *(const int*)(Bs + brow * 128 + s * 64 + boff);
        }
      }
#pragma unroll
      for (int im = 0; im < 2; im++)
#pragma unroll
        for (int in = 0; in < 2; in++)
          acc[im][in] = __builtin_amdgcn_wmma_i32_16x16x64_iu8(
              true, a[im], true, b[in], acc[im][in], false, false);
    }
    __syncthreads();
  }
#pragma unroll
  for (int im = 0; im < 2; im++) {
#pragma unroll
    for (int in = 0; in < 2; in++) {
      int ng = n0 + (nwp * 2 + in) * 16 + lane15;
      float wsc = w2_scale[e * H_DIM + ng];
#pragma unroll
      for (int r = 0; r < 8; r++) {
        int mloc = (mwp * 2 + im) * 16 + r + hi8;
        int slot = m0 + mloc;
        if (slot < count) {
          int token = toks[mloc];
          float y = (float)acc[im][in][r] * ass[mloc] * wsc;
          out[(size_t)token * H_DIM + ng] += gl[mloc] * y;
        }
      }
    }
  }
}

extern "C" void kernel_launch(void* const* d_in, const int* in_sizes, int n_in,
                              void* d_out, int out_size, void* d_ws, size_t ws_size,
                              hipStream_t stream) {
  const float* x         = (const float*)d_in[0];
  const float* gate_w    = (const float*)d_in[1];
  const int*   w13_q     = (const int*)d_in[2];
  const float* w13_scale = (const float*)d_in[3];
  const int*   w2_q      = (const int*)d_in[4];
  const float* w2_scale  = (const float*)d_in[5];
  const float* shared_o  = (const float*)d_in[6];
  float* out = (float*)d_out;

  char* p = (char*)d_ws;
  auto alloc = [&](size_t bytes) {
    char* r = p;
    p += (bytes + 255) & ~(size_t)255;
    return r;
  };
  char*  xq        = (char*)alloc((size_t)T_TOK * H_DIM);
  float* xs        = (float*)alloc((size_t)T_TOK * 4);
  char*  w13p      = (char*)alloc((size_t)E_EXP * N1_DIM * H_DIM);
  char*  w2p       = (char*)alloc((size_t)E_EXP * H_DIM * I_DIM);
  int*   cnt       = (int*)alloc(E_EXP * 4);
  int*   tok_list  = (int*)alloc((size_t)E_EXP * T_TOK * 4);
  float* gate_list = (float*)alloc((size_t)E_EXP * T_TOK * 4);
  float* h_buf     = (float*)alloc((size_t)T_TOK * N1_DIM * 4);
  char*  aq        = (char*)alloc((size_t)T_TOK * I_DIM);
  float* as_       = (float*)alloc((size_t)T_TOK * 4);

  hipMemcpyAsync(out, shared_o, (size_t)T_TOK * H_DIM * sizeof(float),
                 hipMemcpyDeviceToDevice, stream);

  int n4_13 = E_EXP * N1_DIM * H_DIM / 4;
  int n4_2  = E_EXP * H_DIM * I_DIM / 4;
  pack_i32_i8_kernel<<<(n4_13 + 255) / 256, 256, 0, stream>>>(w13_q, (int*)w13p, n4_13);
  pack_i32_i8_kernel<<<(n4_2 + 255) / 256, 256, 0, stream>>>(w2_q, (int*)w2p, n4_2);

  zero_cnt_kernel<<<1, 32, 0, stream>>>(cnt);
  gate_topk_kernel<<<T_TOK / 8, 256, 0, stream>>>(x, gate_w, cnt, tok_list, gate_list);
  quant_x_kernel<<<T_TOK / 8, 256, 0, stream>>>(x, xq, xs);

  for (int e = 0; e < E_EXP; e++) {
    fc1_kernel<<<dim3(N1_DIM / 64, T_TOK / 128), 256, 0, stream>>>(
        xq, xs, w13p, w13_scale, cnt, tok_list, h_buf, e);
    swiglu_quant_kernel<<<T_TOK / 8, 256, 0, stream>>>(h_buf, aq, as_, cnt, e);
    fc2_kernel<<<dim3(H_DIM / 64, T_TOK / 128), 256, 0, stream>>>(
        aq, as_, w2p, w2_scale, cnt, tok_list, gate_list, out, e);
  }
}
